// HFAB_39496519254692
// MI455X (gfx1250) — matrix operand order repaired and checked by reading, not compiled, not run on hardware
//
#include <hip/hip_runtime.h>
#include <hip/hip_bf16.h>

// ---------------------------------------------------------------------------
// MI455X (gfx1250): wave32, WMMA f32_16x16x32_f16, async global->LDS staging,
// pre-swizzled f16 fragment tables for all reused GEMM operands.
// Workspace requirement: ~373 MiB (93,323,264 floats). Layout documented below.
// ---------------------------------------------------------------------------

typedef __attribute__((ext_vector_type(16))) _Float16 v16h;
typedef __attribute__((ext_vector_type(8)))  float    v8f;

union HFrag { _Float16 h[16]; v16h v; };

__device__ __forceinline__ v8f wmma16(const v16h& a, const v16h& b, v8f c) {
  return __builtin_amdgcn_wmma_f32_16x16x32_f16(false, a, false, b, (short)0, c,
                                                false, false);
}

// A-matrix (16x32 f16) K index for (half, vgpr j, halfword t)
__device__ __forceinline__ int a_kidx(int half, int j, int t) {
  return 16 * half + 2 * j + t;
}

__device__ __forceinline__ v16h ldfrag(const _Float16* p) {
  return *reinterpret_cast<const v16h*>(p);
}

// LDS byte offset of a flat pointer to __shared__ (flat-LDS aperture keeps the
// DS offset in addr[31:0], ISA 10.2)
__device__ __forceinline__ unsigned lds_off(const void* p) {
  return (unsigned)(size_t)p;
}

// Async gather of one 16-byte chunk from global memory into LDS (gfx1250).
__device__ __forceinline__ void async_b128_to_lds(unsigned loff,
                                                  const void* gaddr) {
  unsigned long long ga = (unsigned long long)gaddr;
  asm volatile("global_load_async_to_lds_b128 %0, %1, off"
               :: "v"(loff), "v"(ga)
               : "memory");
}
__device__ __forceinline__ void wait_async0() {
  asm volatile("s_wait_asynccnt 0" ::: "memory");
}

// ---------------------------------------------------------------------------
// K0: inv_rms over 128 channels per pixel
// ---------------------------------------------------------------------------
__global__ __launch_bounds__(256) void k0_rms(const float* __restrict__ x,
                                              float* __restrict__ invr) {
  int i = blockIdx.x * 256 + threadIdx.x;
  int b = i >> 16, p = i & 65535;
  float s = 0.f;
  #pragma unroll 8
  for (int c = 0; c < 128; ++c) {
    float v = x[(((size_t)b * 128 + c) << 16) + p];
    s += v * v;
  }
  invr[i] = rsqrtf(s * (1.0f / 128.0f) + 1e-6f);
}

// ---------------------------------------------------------------------------
// DFT tables: Dc = cos(2pi kn/256)/16, Ds = -sin(2pi kn/256)/16 (ortho, fwd)
// ---------------------------------------------------------------------------
__global__ __launch_bounds__(256) void k_dft_init(float* __restrict__ Dc,
                                                  float* __restrict__ Ds) {
  int i = blockIdx.x * 256 + threadIdx.x;            // 65536
  int k = i >> 8, nn = i & 255;
  int mm = (k * nn) & 255;
  float ang = (float)mm * (6.2831853071795864769f / 256.0f);
  Dc[i] = cosf(ang) * (1.0f / 16.0f);
  Ds[i] = -sinf(ang) * (1.0f / 16.0f);
}

// Pre-swizzle DFT tables into f16 A-fragment layout [mt][kc][lane][16]
__global__ __launch_bounds__(256) void k_swzA_dft(
    const float* __restrict__ Dc, const float* __restrict__ Ds,
    _Float16* __restrict__ DcA, _Float16* __restrict__ DsAp,
    _Float16* __restrict__ DsAn) {
  int i = blockIdx.x * 256 + threadIdx.x;            // 65536
  int e = i & 15, f = i >> 4;
  int lane = f & 31, kc = (f >> 5) & 7, mt = f >> 8;
  int j = e >> 1, t = e & 1, m = lane & 15, half = lane >> 4;
  int row = mt * 16 + m;
  int k = kc * 32 + a_kidx(half, j, t);
  float c = Dc[row * 256 + k], s = Ds[row * 256 + k];
  DcA[i] = (_Float16)c;
  DsAp[i] = (_Float16)s;
  DsAn[i] = (_Float16)(-s);
}

// Pre-swizzle DFT tables into f16 B-fragment layout [kc][half][col][16]
__global__ __launch_bounds__(256) void k_swzB_dft(
    const float* __restrict__ Dc, const float* __restrict__ Ds,
    _Float16* __restrict__ DcB, _Float16* __restrict__ DsB) {
  int i = blockIdx.x * 256 + threadIdx.x;            // 65536
  int e = i & 15, f = i >> 4;
  int col = f & 255, half = (f >> 8) & 1, kc = f >> 9;
  int j = e >> 1, t = e & 1;
  int k = kc * 32 + 16 * half + 2 * j + t;
  DcB[i] = (_Float16)Dc[k * 256 + col];
  DsB[i] = (_Float16)Ds[k * 256 + col];
}

// Generic weight swizzle into f16 A-fragment layout, optional per-k scale.
__global__ __launch_bounds__(256) void k_swz_w(
    const float* __restrict__ W, const float* __restrict__ scale, int Kdim,
    int total, _Float16* __restrict__ outT) {
  int i = blockIdx.x * 256 + threadIdx.x;
  if (i >= total) return;
  int e = i & 15, f = i >> 4;
  int lane = f & 31, rest = f >> 5;
  int nkc = Kdim >> 5;
  int kc = rest % nkc, mt = rest / nkc;
  int j = e >> 1, t = e & 1, m = lane & 15, half = lane >> 4;
  int o = mt * 16 + m;
  int k = kc * 32 + a_kidx(half, j, t);
  float v = W[o * Kdim + k];
  if (scale) v *= scale[k];
  outT[i] = (_Float16)v;
}

// ---------------------------------------------------------------------------
// K1: fused norm + QKV (64->64) + gate (32->32, sigmoid) + xf_n staging.
// Input tile async-staged to LDS; weights from pre-swizzled f16 tables.
// ---------------------------------------------------------------------------
__global__ __launch_bounds__(128) void k1_qkv_gate(
    const float* __restrict__ x, const float* __restrict__ nscale,
    const _Float16* __restrict__ wqs, const _Float16* __restrict__ wks,
    const _Float16* __restrict__ wvs, const _Float16* __restrict__ gws,
    const float* __restrict__ invr, float* __restrict__ qhat,
    float* __restrict__ khat, float* __restrict__ vbuf,
    float* __restrict__ xfn, float* __restrict__ comb) {
  __shared__ unsigned int ldsX[4][2048];             // 8 KiB / wave
  const int lane = threadIdx.x & 31;
  const int wid = threadIdx.x >> 5;
  const int tile = blockIdx.x * 4 + wid;             // 16384 pixel tiles
  const int b = tile >> 12;
  const int p0 = (tile & 4095) << 4;
  const int half = lane >> 4, n = lane & 15;
  const int p = p0 + n;
  const float ir = invr[(b << 16) + p];

  // async stage x[b, 0:128, p0:p0+16] -> LDS [c][pix]
  float* tf = (float*)ldsX[wid];
  #pragma unroll
  for (int it = 0; it < 16; ++it) {
    int g = it * 32 + lane;
    int c = g >> 2, ng = (g & 3) * 4;
    const float* gp = x + (((size_t)b * 128 + c) << 16) + p0 + ng;
    async_b128_to_lds(lds_off(&tf[c * 16 + ng]), gp);
  }
  wait_async0();                                     // wave-local region
  const float* tilef = (const float*)ldsX[wid];

  // B fragments from LDS: xs chunks (0..31, 32..63), gate chunk (96..127)
  v16h bs0, bs1, bg;
  #pragma unroll
  for (int sel = 0; sel < 3; ++sel) {
    int cbase = (sel < 2) ? sel * 32 : 96;
    HFrag f;
    #pragma unroll
    for (int j = 0; j < 8; ++j)
      #pragma unroll
      for (int t = 0; t < 2; ++t) {
        int c = cbase + 16 * half + 2 * j + t;
        f.h[2 * j + t] = (_Float16)(tilef[c * 16 + n] * ir);
      }
    if (sel == 0) bs0 = f.v; else if (sel == 1) bs1 = f.v; else bg = f.v;
  }

  v8f accQ[4] = {}, accK[4] = {}, accV[4] = {}, accG[2] = {};
  #pragma unroll
  for (int kc = 0; kc < 2; ++kc) {
    v16h bfr = kc ? bs1 : bs0;
    #pragma unroll
    for (int mt = 0; mt < 4; ++mt) {
      size_t off = (size_t)(((mt * 2 + kc) * 32 + lane) * 16);
      accQ[mt] = wmma16(ldfrag(wqs + off), bfr, accQ[mt]);
      accK[mt] = wmma16(ldfrag(wks + off), bfr, accK[mt]);
      accV[mt] = wmma16(ldfrag(wvs + off), bfr, accV[mt]);
    }
  }
  #pragma unroll
  for (int mt = 0; mt < 2; ++mt) {
    size_t off = (size_t)((mt * 32 + lane) * 16);
    accG[mt] = wmma16(ldfrag(gws + off), bg, accG[mt]);
  }

  #pragma unroll
  for (int mt = 0; mt < 4; ++mt)
    #pragma unroll
    for (int r = 0; r < 8; ++r) {
      int o = mt * 16 + 8 * half + r;
      size_t idx = (((size_t)b * 64 + o) << 16) + p;
      qhat[idx] = accQ[mt][r];
      khat[idx] = accK[mt][r];
      vbuf[idx] = accV[mt][r];
    }
  #pragma unroll
  for (int mt = 0; mt < 2; ++mt)
    #pragma unroll
    for (int r = 0; r < 8; ++r) {
      int o = mt * 16 + 8 * half + r;
      float z = accG[mt][r];
      comb[(((size_t)b * 128 + 96 + o) << 16) + p] = 1.0f / (1.0f + __expf(-z));
    }
  // xf_n = normalized channels 64..95 (read from the LDS tile)
  #pragma unroll
  for (int cc = 0; cc < 16; ++cc) {
    int idx = cc * 32 + lane;
    int cf = idx >> 4, m2 = idx & 15;
    int p2 = p0 + m2;
    float ir2 = invr[(b << 16) + p2];
    xfn[(((size_t)b * 32 + cf) << 16) + p2] =
        tilef[(64 + cf) * 16 + m2] * ir2 * nscale[64 + cf];
  }
}

// ---------------------------------------------------------------------------
// K2: k = elu(dwconv3(khat))+1 ; reduce kv = sum k*v, den = sum k per (b,c)
// ---------------------------------------------------------------------------
__global__ __launch_bounds__(256) void k2_kreduce(
    const float* __restrict__ khat, const float* __restrict__ vbuf,
    const float* __restrict__ dwqk, float* __restrict__ kv,
    float* __restrict__ den) {
  __shared__ float s1[256], s2[256];
  int bc = blockIdx.x;
  int c = bc & 63;
  float w[9];
  #pragma unroll
  for (int i = 0; i < 9; ++i) w[i] = dwqk[c * 9 + i];
  const float* kp = khat + ((size_t)bc << 16);
  const float* vp = vbuf + ((size_t)bc << 16);
  float sk = 0.f, skv = 0.f;
  for (int idx = threadIdx.x; idx < 65536; idx += 256) {
    int y = idx >> 8, xx = idx & 255;
    float acc = 0.f;
    #pragma unroll
    for (int dy = 0; dy < 3; ++dy) {
      int yy = y + dy - 1;
      if (yy < 0 || yy > 255) continue;
      #pragma unroll
      for (int dx = 0; dx < 3; ++dx) {
        int xc = xx + dx - 1;
        if (xc < 0 || xc > 255) continue;
        acc += w[dy * 3 + dx] * kp[yy * 256 + xc];
      }
    }
    float kk = acc > 0.f ? acc + 1.f : __expf(acc);
    sk += kk;
    skv += kk * vp[idx];
  }
  s1[threadIdx.x] = sk; s2[threadIdx.x] = skv;
  __syncthreads();
  for (int st = 128; st > 0; st >>= 1) {
    if (threadIdx.x < st) {
      s1[threadIdx.x] += s1[threadIdx.x + st];
      s2[threadIdx.x] += s2[threadIdx.x + st];
    }
    __syncthreads();
  }
  if (threadIdx.x == 0) { kv[bc] = s2[0]; den[bc] = s1[0] + 1e-6f; }
}

// ---------------------------------------------------------------------------
// K3: q = elu(dwconv3(qhat))+1 ; xs_out = (q*kv/den + v)*(ss+rb[c]) * gate
// ---------------------------------------------------------------------------
__global__ __launch_bounds__(256) void k3_spatial_out(
    const float* __restrict__ qhat, const float* __restrict__ vbuf,
    const float* __restrict__ dwqk, const float* __restrict__ kv,
    const float* __restrict__ den, const float* __restrict__ sscale,
    const float* __restrict__ rbias, float* __restrict__ comb) {
  size_t i = (size_t)blockIdx.x * 256 + threadIdx.x;
  int plane = (int)(i >> 16);
  int b = plane >> 6, c = plane & 63;
  int p = (int)(i & 65535);
  int y = p >> 8, xx = p & 255;
  const float* qp = qhat + ((size_t)plane << 16);
  float acc = 0.f;
  #pragma unroll
  for (int dy = 0; dy < 3; ++dy) {
    int yy = y + dy - 1; if (yy < 0 || yy > 255) continue;
    #pragma unroll
    for (int dx = 0; dx < 3; ++dx) {
      int xc = xx + dx - 1; if (xc < 0 || xc > 255) continue;
      acc += dwqk[c * 9 + dy * 3 + dx] * qp[yy * 256 + xc];
    }
  }
  float q = acc > 0.f ? acc + 1.f : __expf(acc);
  float xs = (q * kv[plane] / den[plane] + vbuf[i]) * (sscale[0] + rbias[c]);
  float g = comb[(((size_t)b * 128 + 96 + (c & 31)) << 16) + p];
  comb[(((size_t)b * 128 + c) << 16) + p] = xs * g;
}

// ---------------------------------------------------------------------------
// cgemm_As: C = (DcA + i*(sa)DsA) * (Br + i*Bi), shared swizzled A tables.
// AfP/AfN carry +sa*Ds and -sa*Ds (host selects). Full complex. (S2, S3)
// ---------------------------------------------------------------------------
__global__ __launch_bounds__(128) void cgemm_As(
    const _Float16* __restrict__ AfC, const _Float16* __restrict__ AfP,
    const _Float16* __restrict__ AfN, const float* __restrict__ Br,
    const float* __restrict__ Bi, int sBp, float* __restrict__ Cr,
    float* __restrict__ Ci, int sCp, const float* __restrict__ cw) {
  const int lane = threadIdx.x & 31;
  const int wid = threadIdx.x >> 5;
  const int plane = blockIdx.y;
  const int tileId = blockIdx.x * 4 + wid;
  const int mTile = tileId >> 2;
  const int m0 = mTile * 16, n0 = (tileId & 3) * 64;
  const int half = lane >> 4, n = lane & 15;

  const float* BrP = Br + (size_t)plane * sBp;
  const float* BiP = Bi + (size_t)plane * sBp;

  v8f cr[4] = {}, ci[4] = {};
  #pragma unroll
  for (int kc = 0; kc < 8; ++kc) {
    size_t aoff = (size_t)(((mTile * 8 + kc) * 32 + lane) * 16);
    v16h far_ = ldfrag(AfC + aoff);
    v16h faip = ldfrag(AfP + aoff);
    v16h fain = ldfrag(AfN + aoff);
    #pragma unroll
    for (int nt = 0; nt < 4; ++nt) {
      int col = n0 + nt * 16 + n;
      HFrag fbr, fbi;
      #pragma unroll
      for (int j = 0; j < 8; ++j)
        #pragma unroll
        for (int t = 0; t < 2; ++t) {
          int k = kc * 32 + 16 * half + 2 * j + t;
          fbr.h[2 * j + t] = (_Float16)BrP[k * 256 + col];
          fbi.h[2 * j + t] = (_Float16)BiP[k * 256 + col];
        }
      cr[nt] = wmma16(far_, fbr.v, cr[nt]);
      cr[nt] = wmma16(fain, fbi.v, cr[nt]);
      ci[nt] = wmma16(far_, fbi.v, ci[nt]);
      ci[nt] = wmma16(faip, fbr.v, ci[nt]);
    }
  }

  if (cw) {  // spectral multiply: (a + i*b*sigma(kx)), sigma=0 at DC/Nyquist
    int cf = plane & 31;
    float a = cw[2 * cf], bb = cw[2 * cf + 1];
    #pragma unroll
    for (int nt = 0; nt < 4; ++nt) {
      int kx = n0 + nt * 16 + n;
      float s = (kx == 0 || kx == 128) ? 0.0f : (kx < 128 ? 1.0f : -1.0f);
      #pragma unroll
      for (int r = 0; r < 8; ++r) {
        float xr = cr[nt][r], xi = ci[nt][r];
        cr[nt][r] = a * xr - bb * s * xi;
        ci[nt][r] = a * xi + bb * s * xr;
      }
    }
  }

  float* CrP = Cr + (size_t)plane * sCp;
  float* CiP = Ci + (size_t)plane * sCp;
  #pragma unroll
  for (int nt = 0; nt < 4; ++nt)
    #pragma unroll
    for (int r = 0; r < 8; ++r) {
      int row = m0 + r + 8 * half;
      int col = n0 + nt * 16 + n;
      CrP[row * 256 + col] = cr[nt][r];
      CiP[row * 256 + col] = ci[nt][r];
    }
}

// ---------------------------------------------------------------------------
// cgemm_Bs: C = (Ar + i*sa*Ai) * (DcB + i*DsB), shared swizzled B tables.
// Ai nullable (S1); Ci nullable (S4: real output only).
// ---------------------------------------------------------------------------
__global__ __launch_bounds__(128) void cgemm_Bs(
    const float* __restrict__ Ar, const float* __restrict__ Ai, float sa,
    int sAp, const _Float16* __restrict__ BfC, const _Float16* __restrict__ BfS,
    float* __restrict__ Cr, float* __restrict__ Ci, int sCp) {
  const int lane = threadIdx.x & 31;
  const int wid = threadIdx.x >> 5;
  const int plane = blockIdx.y;
  const int tileId = blockIdx.x * 4 + wid;
  const int m0 = (tileId >> 2) * 16, n0 = (tileId & 3) * 64;
  const int half = lane >> 4, n = lane & 15, m = n;

  const float* ArP = Ar + (size_t)plane * sAp;
  const float* AiP = Ai ? Ai + (size_t)plane * sAp : nullptr;
  const bool wantCi = (Ci != nullptr);

  v8f cr[4] = {}, ci[4] = {};
  #pragma unroll
  for (int kc = 0; kc < 8; ++kc) {
    HFrag far_, faip, fain;
    #pragma unroll
    for (int j = 0; j < 8; ++j)
      #pragma unroll
      for (int t = 0; t < 2; ++t) {
        int k = kc * 32 + a_kidx(half, j, t);
        far_.h[2 * j + t] = (_Float16)ArP[(m0 + m) * 256 + k];
        if (AiP) {
          float v = AiP[(m0 + m) * 256 + k];
          faip.h[2 * j + t] = (_Float16)(sa * v);
          fain.h[2 * j + t] = (_Float16)(-sa * v);
        }
      }
    #pragma unroll
    for (int nt = 0; nt < 4; ++nt) {
      int col = n0 + nt * 16 + n;
      size_t boff = (size_t)((((kc * 2 + half) * 256 + col) * 16));
      v16h fbr = ldfrag(BfC + boff);
      v16h fbi = ldfrag(BfS + boff);
      cr[nt] = wmma16(far_.v, fbr, cr[nt]);
      if (AiP) cr[nt] = wmma16(fain.v, fbi, cr[nt]);
      if (wantCi) {
        ci[nt] = wmma16(far_.v, fbi, ci[nt]);
        if (AiP) ci[nt] = wmma16(faip.v, fbr, ci[nt]);
      }
    }
  }

  float* CrP = Cr + (size_t)plane * sCp;
  float* CiP = wantCi ? Ci + (size_t)plane * sCp : nullptr;
  #pragma unroll
  for (int nt = 0; nt < 4; ++nt)
    #pragma unroll
    for (int r = 0; r < 8; ++r) {
      int row = m0 + r + 8 * half;
      int col = n0 + nt * 16 + n;
      CrP[row * 256 + col] = cr[nt][r];
      if (wantCi) CiP[row * 256 + col] = ci[nt][r];
    }
}

// ---------------------------------------------------------------------------
// Kf: inv_rms over the 32 freq channels per pixel
// ---------------------------------------------------------------------------
__global__ __launch_bounds__(256) void kf_rms(const float* __restrict__ xfsp,
                                              float* __restrict__ invf) {
  int i = blockIdx.x * 256 + threadIdx.x;
  int b = i >> 16, p = i & 65535;
  float s = 0.f;
  #pragma unroll 8
  for (int c = 0; c < 32; ++c) {
    float v = xfsp[(((size_t)b * 32 + c) << 16) + p];
    s += v * v;
  }
  invf[i] = rsqrtf(s * (1.0f / 32.0f) + 1e-6f);
}

// ---------------------------------------------------------------------------
// S5: freq 1x1 conv (32->32, fnorm folded in table), times gate -> comb[64:96]
// ---------------------------------------------------------------------------
__global__ __launch_bounds__(128) void s5_freq_proj(
    const float* __restrict__ xfsp, const float* __restrict__ invf,
    const _Float16* __restrict__ fws, float* __restrict__ comb) {
  const int lane = threadIdx.x & 31;
  const int wid = threadIdx.x >> 5;
  const int tile = blockIdx.x * 4 + wid;
  const int b = tile >> 12;
  const int p0 = (tile & 4095) << 4;
  const int half = lane >> 4, n = lane & 15;
  const int p = p0 + n;
  const float ir = invf[(b << 16) + p];

  HFrag fb;
  #pragma unroll
  for (int j = 0; j < 8; ++j)
    #pragma unroll
    for (int t = 0; t < 2; ++t) {
      int k = 16 * half + 2 * j + t;
      fb.h[2 * j + t] = (_Float16)(xfsp[(((size_t)b * 32 + k) << 16) + p] * ir);
    }
  v8f acc[2] = {};
  #pragma unroll
  for (int mt = 0; mt < 2; ++mt) {
    size_t off = (size_t)((mt * 32 + lane) * 16);
    acc[mt] = wmma16(ldfrag(fws + off), fb.v, acc[mt]);
  }
  #pragma unroll
  for (int mt = 0; mt < 2; ++mt)
    #pragma unroll
    for (int r = 0; r < 8; ++r) {
      int o = mt * 16 + 8 * half + r;
      float g = comb[(((size_t)b * 128 + 96 + o) << 16) + p];
      comb[(((size_t)b * 128 + 64 + o) << 16) + p] = acc[mt][r] * g;
    }
}

// ---------------------------------------------------------------------------
// K45: fused proj_out GEMM (128->128) + rmsnorm(norm2) + MLP (128->256->128).
// Input tile async-staged to LDS; weights from pre-swizzled f16 tables; the
// 256-ch hidden activation lives only in registers + LDS (never HBM).
// ---------------------------------------------------------------------------
__global__ __launch_bounds__(128) void k45_proj_mlp(
    const float* __restrict__ comb, const _Float16* __restrict__ pws,
    const float* __restrict__ n2s, const _Float16* __restrict__ w1s,
    const _Float16* __restrict__ w2s, float* __restrict__ out) {
  __shared__ unsigned int ldsA[4][2048];             // 8 KiB / wave
  __shared__ unsigned int ldsB[4][1024];             // 4 KiB / wave
  const int lane = threadIdx.x & 31;
  const int wid = threadIdx.x >> 5;
  unsigned int* LA = ldsA[wid];
  unsigned int* LB = ldsB[wid];
  const int tile = blockIdx.x * 4 + wid;
  const int b = tile >> 12;
  const int p0 = (tile & 4095) << 4;
  const int half = lane >> 4, n = lane & 15;
  const int p = p0 + n;

  // async stage comb[b, 0:128, p0:p0+16] -> LDS tile [c][pix] (f32)
  {
    float* tf = (float*)LA;
    #pragma unroll
    for (int it = 0; it < 16; ++it) {
      int g = it * 32 + lane;
      int c = g >> 2, ng = (g & 3) * 4;
      const float* gp = comb + (((size_t)b * 128 + c) << 16) + p0 + ng;
      async_b128_to_lds(lds_off(&tf[c * 16 + ng]), gp);
    }
    wait_async0();
  }
  const float* tilef = (const float*)LA;

  // proj GEMM: M=128 (8 tiles), K=128 (4 chunks), B from LDS
  v8f accP[8] = {};
  #pragma unroll
  for (int kc = 0; kc < 4; ++kc) {
    HFrag fb;
    #pragma unroll
    for (int j = 0; j < 8; ++j)
      #pragma unroll
      for (int t = 0; t < 2; ++t) {
        int c = kc * 32 + 16 * half + 2 * j + t;
        fb.h[2 * j + t] = (_Float16)tilef[c * 16 + n];
      }
    #pragma unroll
    for (int mt = 0; mt < 8; ++mt) {
      size_t off = (size_t)(((mt * 4 + kc) * 32 + lane) * 16);
      accP[mt] = wmma16(ldfrag(pws + off), fb.v, accP[mt]);
    }
  }

  // rmsnorm over channels: lanes L and L+16 hold the same pixel column
  float ssq = 0.f;
  #pragma unroll
  for (int mt = 0; mt < 8; ++mt)
    #pragma unroll
    for (int r = 0; r < 8; ++r) ssq += accP[mt][r] * accP[mt][r];
  ssq += __shfl_xor(ssq, 16);
  float inv2 = rsqrtf(ssq * (1.0f / 128.0f) + 1e-6f);

  // y2 = norm2(proj) as f16 pairs into ldsB [o/2][n]
  #pragma unroll
  for (int mt = 0; mt < 8; ++mt)
    #pragma unroll
    for (int r = 0; r < 8; ++r) {
      int o = mt * 16 + 8 * half + r;
      float v = accP[mt][r] * inv2 * n2s[o];
      ((unsigned short*)LB)[((o >> 1) * 16 + n) * 2 + (o & 1)] =
          __builtin_bit_cast(unsigned short, (_Float16)v);
    }
  __syncthreads();

  // MLP GEMM1 in two passes of 8 M-tiles (keeps VGPR pressure low)
  #pragma unroll
  for (int pass = 0; pass < 2; ++pass) {
    v8f accH[8] = {};
    #pragma unroll
    for (int kc = 0; kc < 4; ++kc) {
      HFrag fb;
      #pragma unroll
      for (int j = 0; j < 8; ++j) {
        unsigned int d = LB[(kc * 16 + 8 * half + j) * 16 + n];
        fb.h[2 * j] = __builtin_bit_cast(_Float16, (unsigned short)(d & 0xffff));
        fb.h[2 * j + 1] = __builtin_bit_cast(_Float16, (unsigned short)(d >> 16));
      }
      #pragma unroll
      for (int mt = 0; mt < 8; ++mt) {
        size_t off = (size_t)((((pass * 8 + mt) * 4 + kc) * 32 + lane) * 16);
        accH[mt] = wmma16(ldfrag(w1s + off), fb.v, accH[mt]);
      }
    }
    // exact gelu, stage hidden rows [pass*128, pass*128+128) into ldsA
    #pragma unroll
    for (int mt = 0; mt < 8; ++mt)
      #pragma unroll
      for (int r = 0; r < 8; ++r) {
        int o = (pass * 8 + mt) * 16 + 8 * half + r;
        float z = accH[mt][r];
        float g = 0.5f * z * (1.0f + erff(z * 0.70710678f));
        ((unsigned short*)LA)[((o >> 1) * 16 + n) * 2 + (o & 1)] =
            __builtin_bit_cast(unsigned short, (_Float16)g);
      }
  }
  __syncthreads();

  // MLP GEMM2: out = w2 * hidden ; M=128 (8 tiles), K=256 (8 chunks)
  v8f accO[8] = {};
  #pragma unroll
  for (int kc = 0; kc < 8; ++kc) {
    HFrag fb;
    #pragma unroll
    for (int j = 0; j < 8; ++j) {
      unsigned int d = LA[(kc * 16 + 8 * half + j) * 16 + n];
      fb.h[2 * j] = __builtin_bit_cast(_Float16, (unsigned short)(d & 0xffff));
      fb.h[2 * j + 1] = __builtin_bit_cast(_Float16, (unsigned short)(d >> 16));
    }
    #pragma unroll
    for (int mt = 0; mt < 8; ++mt) {
      size_t off = (size_t)(((mt * 8 + kc) * 32 + lane) * 16);
      accO[mt] = wmma16(ldfrag(w2s + off), fb.v, accO[mt]);
    }
  }
  #pragma unroll
  for (int mt = 0; mt < 8; ++mt)
    #pragma unroll
    for (int r = 0; r < 8; ++r) {
      int o = mt * 16 + 8 * half + r;
      out[(((size_t)b * 128 + o) << 16) + p] = accO[mt][r];
    }
}

// ---------------------------------------------------------------------------
// K6: depthwise 3x3 (mlp_dw) + residual
// ---------------------------------------------------------------------------
__global__ __launch_bounds__(256) void k6_dw_residual(
    const float* __restrict__ mo, const float* __restrict__ dw,
    const float* __restrict__ x, float* __restrict__ out) {
  size_t i = (size_t)blockIdx.x * 256 + threadIdx.x;
  int plane = (int)(i >> 16);
  int c = plane & 127;
  int p = (int)(i & 65535);
  int y = p >> 8, xx = p & 255;
  const float* mp = mo + ((size_t)plane << 16);
  __builtin_prefetch(mp + ((y + 2 < 256 ? y + 2 : y) * 256) + xx, 0, 1);
  float acc = 0.f;
  #pragma unroll
  for (int dy = 0; dy < 3; ++dy) {
    int yy = y + dy - 1; if (yy < 0 || yy > 255) continue;
    #pragma unroll
    for (int dx = 0; dx < 3; ++dx) {
      int xc = xx + dx - 1; if (xc < 0 || xc > 255) continue;
      acc += dw[c * 9 + dy * 3 + dx] * mp[yy * 256 + xc];
    }
  }
  out[i] = acc + x[i];
}

// ---------------------------------------------------------------------------
// Host-side orchestration
// ---------------------------------------------------------------------------
extern "C" void kernel_launch(void* const* d_in, const int* in_sizes, int n_in,
                              void* d_out, int out_size, void* d_ws,
                              size_t ws_size, hipStream_t stream) {
  (void)in_sizes; (void)n_in; (void)out_size; (void)ws_size;
  const float* x    = (const float*)d_in[0];
  const float* n1s  = (const float*)d_in[1];
  const float* wq   = (const float*)d_in[2];
  const float* wk   = (const float*)d_in[3];
  const float* wv   = (const float*)d_in[4];
  const float* dwqk = (const float*)d_in[5];
  const float* rb   = (const float*)d_in[6];
  const float* ss   = (const float*)d_in[7];
  const float* cwt  = (const float*)d_in[8];
  const float* fns  = (const float*)d_in[9];
  const float* fw   = (const float*)d_in[10];
  const float* gw   = (const float*)d_in[11];
  const float* pw   = (const float*)d_in[12];
  const float* n2s  = (const float*)d_in[13];
  const float* w1   = (const float*)d_in[14];
  const float* w2   = (const float*)d_in[15];
  const float* mdw  = (const float*)d_in[16];
  float* out = (float*)d_out;
  float* w = (float*)d_ws;

  // workspace layout (floats); total 93,323,264 floats (~373 MiB)
  const size_t O_RMS1 = 0;
  const size_t O_INVF = 262144;
  const size_t O_KV   = 524288;
  const size_t O_DEN  = 524544;
  const size_t O_DC   = 524800;
  const size_t O_DS   = 590336;
  const size_t O_F16  = 655872;       // f16 fragment tables start here
  const size_t O_QHAT = 1048576;      // 16,777,216 floats
  const size_t O_KHAT = 17825792;     // 16,777,216
  const size_t O_V    = 34603008;     // 16,777,216
  const size_t O_XFN  = 51380224;     // 8,388,608
  const size_t O_COMB = 59768832;     // 33,554,432
  // aliases (phases do not overlap in time):
  const size_t O_F0R = O_QHAT, O_F0I = O_QHAT + 8388608;
  const size_t O_F1R = O_KHAT, O_F1I = O_KHAT + 8388608;
  const size_t O_XFSP = O_V;
  const size_t O_MLP = O_QHAT;

  // f16 fragment tables (offsets in f16 elements within O_F16 region)
  _Float16* hw = (_Float16*)(w + O_F16);
  _Float16* DcA  = hw;                 // 65536
  _Float16* DsAp = hw + 65536;         // 65536
  _Float16* DsAn = hw + 131072;        // 65536
  _Float16* DcB  = hw + 196608;        // 65536
  _Float16* DsB  = hw + 262144;        // 65536
  _Float16* wqs  = hw + 327680;        // 4096
  _Float16* wks  = hw + 331776;        // 4096
  _Float16* wvs  = hw + 335872;        // 4096
  _Float16* gws  = hw + 339968;        // 1024
  _Float16* fws  = hw + 340992;        // 1024
  _Float16* pws  = hw + 342016;        // 16384
  _Float16* w1s  = hw + 358400;        // 32768
  _Float16* w2s  = hw + 391168;        // 32768  (ends 423936 < 785408 cap)

  // ---- one-time (per launch) table construction ----
  k_dft_init<<<256, 256, 0, stream>>>(w + O_DC, w + O_DS);
  k_swzA_dft<<<256, 256, 0, stream>>>(w + O_DC, w + O_DS, DcA, DsAp, DsAn);
  k_swzB_dft<<<256, 256, 0, stream>>>(w + O_DC, w + O_DS, DcB, DsB);
  k_swz_w<<<16, 256, 0, stream>>>(wq, n1s, 64, 4096, wqs);
  k_swz_w<<<16, 256, 0, stream>>>(wk, n1s, 64, 4096, wks);
  k_swz_w<<<16, 256, 0, stream>>>(wv, n1s, 64, 4096, wvs);
  k_swz_w<<<4, 256, 0, stream>>>(gw, n1s + 96, 32, 1024, gws);
  k_swz_w<<<4, 256, 0, stream>>>(fw, fns, 32, 1024, fws);
  k_swz_w<<<64, 256, 0, stream>>>(pw, nullptr, 128, 16384, pws);
  k_swz_w<<<128, 256, 0, stream>>>(w1, nullptr, 128, 32768, w1s);
  k_swz_w<<<128, 256, 0, stream>>>(w2, nullptr, 256, 32768, w2s);

  // ---- main pipeline ----
  k0_rms<<<1024, 256, 0, stream>>>(x, w + O_RMS1);
  k1_qkv_gate<<<4096, 128, 0, stream>>>(x, n1s, wqs, wks, wvs, gws,
                                        w + O_RMS1, w + O_QHAT, w + O_KHAT,
                                        w + O_V, w + O_XFN, w + O_COMB);
  k2_kreduce<<<256, 256, 0, stream>>>(w + O_KHAT, w + O_V, dwqk, w + O_KV,
                                      w + O_DEN);
  k3_spatial_out<<<65536, 256, 0, stream>>>(w + O_QHAT, w + O_V, dwqk,
                                            w + O_KV, w + O_DEN, ss, rb,
                                            w + O_COMB);

  // spectral branch as DFT GEMMs (256x256 per plane, 128 planes)
  dim3 cg(16, 128);
  // S1: T1 = x * (Dc + i Ds)
  cgemm_Bs<<<cg, 128, 0, stream>>>(w + O_XFN, nullptr, 1.f, 65536, DcB, DsB,
                                   w + O_F0R, w + O_F0I, 65536);
  // S2: X = (Dc + i Ds) * T1, then spectral multiply (a + i b sigma)
  cgemm_As<<<cg, 128, 0, stream>>>(DcA, DsAp, DsAn, w + O_F0R, w + O_F0I,
                                   65536, w + O_F1R, w + O_F1I, 65536, cwt);
  // S3: U = (Dc - i Ds) * Y   (sa=-1 -> swap +/- Ds tables)
  cgemm_As<<<cg, 128, 0, stream>>>(DcA, DsAn, DsAp, w + O_F1R, w + O_F1I,
                                   65536, w + O_F0R, w + O_F0I, 65536,
                                   nullptr);
  // S4: y = Ur*Dc + Ui*Ds (real output only, sa=-1)
  cgemm_Bs<<<cg, 128, 0, stream>>>(w + O_F0R, w + O_F0I, -1.f, 65536, DcB,
                                   DsB, w + O_XFSP, nullptr, 65536);

  kf_rms<<<1024, 256, 0, stream>>>(w + O_XFSP, w + O_INVF);
  s5_freq_proj<<<4096, 128, 0, stream>>>(w + O_XFSP, w + O_INVF, fws,
                                         w + O_COMB);

  k45_proj_mlp<<<4096, 128, 0, stream>>>(w + O_COMB, pws, n2s, w1s, w2s,
                                         w + O_MLP);
  k6_dw_residual<<<131072, 256, 0, stream>>>(w + O_MLP, mdw, x, out);
}
